// Attention_19129784336491
// MI455X (gfx1250) — compile-verified, matrix operand order checked by reading, not run on hardware
//
#include <hip/hip_runtime.h>
#include <math.h>

typedef __bf16 bf16;
typedef __attribute__((ext_vector_type(8)))  bf16  v8bf;
typedef __attribute__((ext_vector_type(16))) bf16  v16bf;
typedef __attribute__((ext_vector_type(8)))  float v8f;

#define D_MODEL 1024
#define SEQ     4096
#define BATCH   8
#define NHEADS  16
#define M_ROWS  (BATCH * SEQ)   // 32768
#define LN_EPS  1e-6f

// ---------------------------------------------------------------------------
// Elementwise cast kernels
// ---------------------------------------------------------------------------
__global__ void k_cast_bf16(const float* __restrict__ a, bf16* __restrict__ o, long n) {
  long i  = (long)blockIdx.x * blockDim.x + threadIdx.x;
  long st = (long)gridDim.x * blockDim.x;
  for (; i < n; i += st) o[i] = (bf16)a[i];
}

// u = p_av (broadcast over seq) * v, cast to bf16
__global__ void k_umul_cast(const float* __restrict__ v, const float* __restrict__ pav,
                            bf16* __restrict__ o, long n) {
  long i  = (long)blockIdx.x * blockDim.x + threadIdx.x;
  long st = (long)gridDim.x * blockDim.x;
  for (; i < n; i += st) {
    long b = i >> 22;              // / (SEQ*D_MODEL)
    int  d = (int)(i & (D_MODEL - 1));
    o[i] = (bf16)(v[i] * pav[b * D_MODEL + d]);
  }
}

// (r + q) cast to bf16
__global__ void k_add_cast(const float* __restrict__ r, const float* __restrict__ q,
                           bf16* __restrict__ o, long n) {
  long i  = (long)blockIdx.x * blockDim.x + threadIdx.x;
  long st = (long)gridDim.x * blockDim.x;
  for (; i < n; i += st) o[i] = (bf16)(r[i] + q[i]);
}

// W [K,N] f32 row-major -> Wt [N,K] bf16 row-major (K-contiguous for GEMM B tiles)
__global__ void k_transpose_cast(const float* __restrict__ W, bf16* __restrict__ Wt) {
  __shared__ float t[32][33];
  int k0 = blockIdx.y * 32, n0 = blockIdx.x * 32;
  t[threadIdx.y][threadIdx.x] = W[(size_t)(k0 + threadIdx.y) * D_MODEL + n0 + threadIdx.x];
  __syncthreads();
  Wt[(size_t)(n0 + threadIdx.y) * D_MODEL + k0 + threadIdx.x] = (bf16)t[threadIdx.x][threadIdx.y];
}

// ---------------------------------------------------------------------------
// bf16 WMMA GEMM:  C[M_ROWS,1024] = A[M_ROWS,1024] @ Wt^T + bias
// A row-major bf16 (K-contig), Wt row-major bf16 [N,K] (K-contig).
// Block tile 128x128, 128 threads = 4 waves (2 in M, 2 in N).
// Each wave owns a 64x64 sub-tile = 4x4 fragments of v_wmma_f32_16x16x32_bf16,
// K stepped by 32: 16 WMMAs per 16 ds_load_b128 (1.0 LDS load / WMMA).
// ---------------------------------------------------------------------------
__global__ __launch_bounds__(128)
void k_gemm_bf16(const bf16* __restrict__ A, const bf16* __restrict__ Wt,
                 const float* __restrict__ bias, float* __restrict__ C) {
  const int K = D_MODEL, N = D_MODEL;
  __shared__ bf16 At[128 * 32];
  __shared__ bf16 Bt[128 * 32];

  const int tid  = threadIdx.x;
  const int lane = tid & 31;
  const int w    = tid >> 5;       // 0..3
  const int wm   = w & 1;          // wave M index -> 64 rows each
  const int wn   = w >> 1;         // wave N index -> 64 cols each
  const int m0   = blockIdx.y * 128;
  const int n0   = blockIdx.x * 128;

  const int ml    = lane & 15;     // row/col within a 16-wide fragment
  const int khalf = lane >> 4;     // ISA: lanes 16..31 hold K 8..15 / 24..31
  const int k0sel = khalf * 8;

  v8f acc[4][4];
#pragma unroll
  for (int i = 0; i < 4; ++i)
#pragma unroll
    for (int j = 0; j < 4; ++j) acc[i][j] = {};

  // each thread stages one full 64B row-slice of the A tile and of the B tile
  const bf16* ga = A  + (size_t)(m0 + tid) * K;
  const bf16* gb = Wt + (size_t)(n0 + tid) * K;

  for (int kb = 0; kb < K; kb += 32) {
    // ---- global -> regs
    const int4* pa = (const int4*)(ga + kb);
    int4 a0 = pa[0], a1 = pa[1], a2 = pa[2], a3 = pa[3];
    const int4* pb = (const int4*)(gb + kb);
    int4 b0 = pb[0], b1 = pb[1], b2 = pb[2], b3 = pb[3];

    // prefetch next K-slice (global_prefetch_b8)
    if (kb + 32 < K) {
      __builtin_prefetch(ga + kb + 32, 0, 0);
      __builtin_prefetch(gb + kb + 32, 0, 0);
    }

    __syncthreads();
    int4* la = (int4*)(At + tid * 32);
    la[0] = a0; la[1] = a1; la[2] = a2; la[3] = a3;
    int4* lb = (int4*)(Bt + tid * 32);
    lb[0] = b0; lb[1] = b1; lb[2] = b2; lb[3] = b3;
    __syncthreads();

    // ---- LDS -> fragments (two ds_read_b128 per fragment)
    v16bf af[4], bfm[4];
#pragma unroll
    for (int mt = 0; mt < 4; ++mt) {
      int row = wm * 64 + mt * 16 + ml;
      v8bf lo = *(const v8bf*)(At + row * 32 + k0sel);
      v8bf hi = *(const v8bf*)(At + row * 32 + k0sel + 16);
      af[mt]  = __builtin_shufflevector(lo, hi, 0,1,2,3,4,5,6,7,8,9,10,11,12,13,14,15);
    }
#pragma unroll
    for (int nt = 0; nt < 4; ++nt) {
      int rown = wn * 64 + nt * 16 + ml;
      v8bf lo = *(const v8bf*)(Bt + rown * 32 + k0sel);
      v8bf hi = *(const v8bf*)(Bt + rown * 32 + k0sel + 16);
      bfm[nt] = __builtin_shufflevector(lo, hi, 0,1,2,3,4,5,6,7,8,9,10,11,12,13,14,15);
    }

#pragma unroll
    for (int mt = 0; mt < 4; ++mt)
#pragma unroll
      for (int nt = 0; nt < 4; ++nt)
        acc[mt][nt] = __builtin_amdgcn_wmma_f32_16x16x32_bf16(
            false, af[mt], false, bfm[nt], (short)0, acc[mt][nt], false, false);
  }

  // ---- epilogue: D layout lane = n + 16*(m>>3), vgpr j = m&7
#pragma unroll
  for (int mt = 0; mt < 4; ++mt) {
    int mg = m0 + wm * 64 + mt * 16 + khalf * 8;
#pragma unroll
    for (int nt = 0; nt < 4; ++nt) {
      int   ng = n0 + wn * 64 + nt * 16 + ml;
      float bs = bias[ng];
#pragma unroll
      for (int j = 0; j < 8; ++j)
        C[(size_t)(mg + j) * N + ng] = acc[mt][nt][j] + bs;
    }
  }
}

// ---------------------------------------------------------------------------
// score[b,h,s] = (sum_d Arow[d] * (mul? mul[b,d]:1) * Ws[d,h]) * scale + biasH[h] + mask[b,s]
// 16 seq-rows per block; thread (sl,h) does one (row,head) dot of length 1024.
// ---------------------------------------------------------------------------
__global__ __launch_bounds__(256)
void k_score(const float* __restrict__ Arows, const float* __restrict__ mul,
             const float* __restrict__ Ws, const float* __restrict__ biasH,
             const float* __restrict__ mask, float* __restrict__ sc, float scale) {
  int    t   = threadIdx.x;
  int    sl  = t >> 4, hh = t & 15;
  size_t row = (size_t)blockIdx.x * 16 + sl;
  int    b   = (int)(row >> 12);     // / SEQ
  const float* ap = Arows + row * D_MODEL;
  float acc = 0.f;
  if (mul) {
    const float* mp = mul + (size_t)b * D_MODEL;
    for (int d = 0; d < D_MODEL; ++d) acc += ap[d] * mp[d] * Ws[d * NHEADS + hh];
  } else {
    for (int d = 0; d < D_MODEL; ++d) acc += ap[d] * Ws[d * NHEADS + hh];
  }
  int s = (int)(row & (SEQ - 1));
  sc[((size_t)b * NHEADS + hh) * SEQ + s] = acc * scale + biasH[hh] + mask[(size_t)b * SEQ + s];
}

// ---------------------------------------------------------------------------
// In-place row softmax over S=4096, one block per (b,h) row.
// ---------------------------------------------------------------------------
__global__ __launch_bounds__(256)
void k_softmax(float* __restrict__ sc) {
  float* p = sc + (size_t)blockIdx.x * SEQ;
  int t = threadIdx.x;
  float v[16], m = -1e30f;
#pragma unroll
  for (int i = 0; i < 16; ++i) { v[i] = p[t + i * 256]; m = fmaxf(m, v[i]); }
  __shared__ float red[256];
  red[t] = m; __syncthreads();
  for (int k = 128; k > 0; k >>= 1) { if (t < k) red[t] = fmaxf(red[t], red[t + k]); __syncthreads(); }
  m = red[0]; __syncthreads();
  float sum = 0.f;
#pragma unroll
  for (int i = 0; i < 16; ++i) { v[i] = __expf(v[i] - m); sum += v[i]; }
  red[t] = sum; __syncthreads();
  for (int k = 128; k > 0; k >>= 1) { if (t < k) red[t] += red[t + k]; __syncthreads(); }
  float inv = 1.f / red[0];
#pragma unroll
  for (int i = 0; i < 16; ++i) p[t + i * 256] = v[i] * inv;
}

// ---------------------------------------------------------------------------
// pooled[b,d] = (sum_s wts[b, d>>6, s] * src[b,s,d]) * (mul? mul[b,d]:1)
// ---------------------------------------------------------------------------
__global__ __launch_bounds__(256)
void k_pool(const float* __restrict__ src, const float* __restrict__ wts,
            const float* __restrict__ mul, float* __restrict__ outv) {
  int b = blockIdx.y;
  int d = blockIdx.x * 256 + threadIdx.x;
  int hh = d >> 6;
  const float* wp = wts + ((size_t)b * NHEADS + hh) * SEQ;
  const float* sp = src + (size_t)b * SEQ * D_MODEL + d;
  float acc = 0.f;
  for (int s = 0; s < SEQ; ++s) acc += wp[s] * sp[(size_t)s * D_MODEL];
  if (mul) acc *= mul[(size_t)b * D_MODEL + d];
  outv[(size_t)b * D_MODEL + d] = acc;
}

// ---------------------------------------------------------------------------
// LayerNorm(x + t) * g + b  -> out ; one block per (b,s) row
// ---------------------------------------------------------------------------
__global__ __launch_bounds__(256)
void k_layernorm(const float* __restrict__ x, const float* __restrict__ tbuf,
                 const float* __restrict__ g, const float* __restrict__ bta,
                 float* __restrict__ out) {
  size_t row = blockIdx.x;
  const float* xp = x + row * D_MODEL;
  const float* tp = tbuf + row * D_MODEL;
  int t = threadIdx.x;
  float y[4], s = 0.f, s2 = 0.f;
#pragma unroll
  for (int i = 0; i < 4; ++i) {
    int d = t + i * 256;
    float v = xp[d] + tp[d];
    y[i] = v; s += v; s2 += v * v;
  }
  __shared__ float r1[256], r2[256];
  r1[t] = s; r2[t] = s2; __syncthreads();
  for (int k = 128; k > 0; k >>= 1) {
    if (t < k) { r1[t] += r1[t + k]; r2[t] += r2[t + k]; }
    __syncthreads();
  }
  float mu  = r1[0] * (1.f / D_MODEL);
  float var = r2[0] * (1.f / D_MODEL) - mu * mu;
  float rs  = rsqrtf(var + LN_EPS);
#pragma unroll
  for (int i = 0; i < 4; ++i) {
    int d = t + i * 256;
    out[row * D_MODEL + d] = (y[i] - mu) * rs * g[d] + bta[d];
  }
}

// ---------------------------------------------------------------------------
extern "C" void kernel_launch(void* const* d_in, const int* in_sizes, int n_in,
                              void* d_out, int out_size, void* d_ws, size_t ws_size,
                              hipStream_t stream) {
  const float* x    = (const float*)d_in[0];
  const float* mask = (const float*)d_in[1];
  const float* Wq   = (const float*)d_in[2];
  const float* bq   = (const float*)d_in[3];
  const float* Wk   = (const float*)d_in[4];
  const float* bk   = (const float*)d_in[5];
  const float* Wv   = (const float*)d_in[6];
  const float* bv   = (const float*)d_in[7];
  const float* Wa   = (const float*)d_in[8];
  const float* ba   = (const float*)d_in[9];
  const float* Wb   = (const float*)d_in[10];
  const float* bb   = (const float*)d_in[11];
  const float* Wu   = (const float*)d_in[12];
  const float* bu   = (const float*)d_in[13];
  const float* Wo   = (const float*)d_in[14];
  const float* bo   = (const float*)d_in[15];
  const float* lng  = (const float*)d_in[16];
  const float* lnb  = (const float*)d_in[17];
  float* out = (float*)d_out;

  const long NEL = (long)M_ROWS * D_MODEL;       // 33,554,432
  char* w = (char*)d_ws;
  bf16* xb  = (bf16*)w;  w += (size_t)NEL * 2;   // bf16 x (reused for u and r+q)
  bf16* Wtq = (bf16*)w;  w += (size_t)D_MODEL * D_MODEL * 2;
  bf16* Wtk = (bf16*)w;  w += (size_t)D_MODEL * D_MODEL * 2;
  bf16* Wtv = (bf16*)w;  w += (size_t)D_MODEL * D_MODEL * 2;
  bf16* Wtu = (bf16*)w;  w += (size_t)D_MODEL * D_MODEL * 2;
  bf16* Wto = (bf16*)w;  w += (size_t)D_MODEL * D_MODEL * 2;
  float* qf = (float*)w; w += (size_t)NEL * 4;
  float* kf = (float*)w; w += (size_t)NEL * 4;   // reused as pre-LN activation
  float* vf = (float*)w; w += (size_t)NEL * 4;
  float* rf = (float*)w; w += (size_t)NEL * 4;
  float* sc = (float*)w; w += (size_t)BATCH * NHEADS * SEQ * 4;
  float* qav = (float*)w; w += (size_t)BATCH * D_MODEL * 4;
  float* pav = (float*)w; w += (size_t)BATCH * D_MODEL * 4;
  (void)n_in; (void)in_sizes; (void)out_size; (void)ws_size;

  const float scale = 0.125f;                    // (D/H)^-0.5 = 1/8
  const dim3 gGemm(D_MODEL / 128, M_ROWS / 128); // (8, 256)
  const dim3 gT(32, 32), bT(32, 32);

  // 1) casts
  k_cast_bf16<<<2048, 256, 0, stream>>>(x, xb, NEL);
  k_transpose_cast<<<gT, bT, 0, stream>>>(Wq, Wtq);
  k_transpose_cast<<<gT, bT, 0, stream>>>(Wk, Wtk);
  k_transpose_cast<<<gT, bT, 0, stream>>>(Wv, Wtv);
  k_transpose_cast<<<gT, bT, 0, stream>>>(Wu, Wtu);
  k_transpose_cast<<<gT, bT, 0, stream>>>(Wo, Wto);

  // 2) q, k, v projections (WMMA)
  k_gemm_bf16<<<gGemm, 128, 0, stream>>>(xb, Wtq, bq, qf);
  k_gemm_bf16<<<gGemm, 128, 0, stream>>>(xb, Wtk, bk, kf);
  k_gemm_bf16<<<gGemm, 128, 0, stream>>>(xb, Wtv, bv, vf);

  // 3) alpha pooling: q_av
  k_score<<<M_ROWS / 16, 256, 0, stream>>>(qf, nullptr, Wa, ba, mask, sc, scale);
  k_softmax<<<BATCH * NHEADS, 256, 0, stream>>>(sc);
  k_pool<<<dim3(D_MODEL / 256, BATCH), 256, 0, stream>>>(qf, sc, nullptr, qav);

  // 4) beta pooling of p = k * q_av: p_av (p fused on the fly)
  k_score<<<M_ROWS / 16, 256, 0, stream>>>(kf, qav, Wb, bb, mask, sc, scale);
  k_softmax<<<BATCH * NHEADS, 256, 0, stream>>>(sc);
  k_pool<<<dim3(D_MODEL / 256, BATCH), 256, 0, stream>>>(kf, sc, qav, pav);

  // 5) u = p_av * v ; r = u @ Wu + bu (WMMA)
  k_umul_cast<<<2048, 256, 0, stream>>>(vf, pav, xb, NEL);
  k_gemm_bf16<<<gGemm, 128, 0, stream>>>(xb, Wtu, bu, rf);

  // 6) attn = (r + q) @ Wo + bo (WMMA) ; write into kf (reused)
  k_add_cast<<<2048, 256, 0, stream>>>(rf, qf, xb, NEL);
  k_gemm_bf16<<<gGemm, 128, 0, stream>>>(xb, Wto, bo, kf);

  // 7) residual + LayerNorm -> out
  k_layernorm<<<M_ROWS, 256, 0, stream>>>(x, kf, lng, lnb, out);
}
